// CosineSimilarity_71502615544488
// MI455X (gfx1250) — compile-verified, matrix operand order checked by reading, not run on hardware
//
#include <hip/hip_runtime.h>
#include <hip/hip_bf16.h>

typedef __attribute__((ext_vector_type(16))) __bf16 v16bf;
typedef __attribute__((ext_vector_type(8)))  __bf16 bf16x8;
typedef __attribute__((ext_vector_type(8)))  float  v8f;

#define N_ROWS 32768
#define DIM    2048
#define NEXP   64
#define EPSV   1e-8f

// workspace layout (bytes)
#define WS_EK_HI   0
#define WS_EK_LO   (NEXP*DIM*2)
#define WS_EKT_HI  (2*NEXP*DIM*2)
#define WS_EKT_LO  (3*NEXP*DIM*2)
#define WS_ENORM   (4*NEXP*DIM*2)

// ---------------- prep: split expert keys into bf16 hi/lo, plus transposed copy ----------------
__global__ void prep_split_kernel(const float* __restrict__ ek,
                                  __bf16* __restrict__ ek_hi, __bf16* __restrict__ ek_lo,
                                  __bf16* __restrict__ ekt_hi, __bf16* __restrict__ ekt_lo) {
    int idx = blockIdx.x * blockDim.x + threadIdx.x;   // 0 .. 64*2048-1
    int e = idx >> 11;
    int d = idx & (DIM - 1);
    float x = ek[idx];
    __bf16 h = (__bf16)x;
    __bf16 l = (__bf16)(x - (float)h);
    ek_hi[idx] = h;
    ek_lo[idx] = l;
    ekt_hi[d * NEXP + e] = h;
    ekt_lo[d * NEXP + e] = l;
}

// ---------------- prep: expert norms ----------------
__global__ void enorm_kernel(const float* __restrict__ ek, float* __restrict__ enorm) {
    __shared__ float red[256];
    int e = blockIdx.x;
    const float* row = ek + (size_t)e * DIM;
    float s = 0.f;
    for (int d = threadIdx.x; d < DIM; d += 256) { float v = row[d]; s += v * v; }
    red[threadIdx.x] = s;
    __syncthreads();
    for (int off = 128; off > 0; off >>= 1) {
        if (threadIdx.x < off) red[threadIdx.x] += red[threadIdx.x + off];
        __syncthreads();
    }
    if (threadIdx.x == 0) enorm[e] = sqrtf(red[0]);
}

// ---------------- main fused kernel: 1 wave = 16 rows ----------------
__launch_bounds__(256)
__global__ void moe_cosine_kernel(const float* __restrict__ z,
                                  const __bf16* __restrict__ ek_hi, const __bf16* __restrict__ ek_lo,
                                  const __bf16* __restrict__ ekt_hi, const __bf16* __restrict__ ekt_lo,
                                  const float* __restrict__ enorm,
                                  float* __restrict__ sim_out, float* __restrict__ wexp_out) {
    const int tid  = threadIdx.x;
    const int lane = tid & 31;
    const int wave = tid >> 5;           // 0..7
    const int half = (lane >> 4) & 1;    // K-half select per WMMA A/B layout
    const int mcol = lane & 15;          // row (A) / column (B,C) index
    const int row_base = (blockIdx.x * 8 + wave) * 16;

    __shared__ float lds_w[8][16][68];   // sim, then softmax weights; 68 = pad vs bank conflicts

    // ================= GEMM1: dots[16x64] + row norms =================
    v8f acc[4];
    #pragma unroll
    for (int t = 0; t < 4; ++t) acc[t] = (v8f){0.f,0.f,0.f,0.f,0.f,0.f,0.f,0.f};
    float nsq = 0.f;

    const float* zrow = z + (size_t)(row_base + mcol) * DIM;

    for (int k0 = 0; k0 < DIM; k0 += 32) {
        const int kb = k0 + half * 8;
        __builtin_prefetch(zrow + kb + 64, 0, 1);   // global_prefetch_b8, next tiles
        float4 a0 = *(const float4*)(zrow + kb);
        float4 a1 = *(const float4*)(zrow + kb + 4);
        float4 a2 = *(const float4*)(zrow + kb + 16);
        float4 a3 = *(const float4*)(zrow + kb + 20);
        float af[16] = {a0.x,a0.y,a0.z,a0.w, a1.x,a1.y,a1.z,a1.w,
                        a2.x,a2.y,a2.z,a2.w, a3.x,a3.y,a3.z,a3.w};
        v16bf ah, al;
        #pragma unroll
        for (int i = 0; i < 16; ++i) {
            float x = af[i];
            nsq += x * x;
            __bf16 h = (__bf16)x;
            ah[i] = h;
            al[i] = (__bf16)(x - (float)h);
        }
        #pragma unroll
        for (int t = 0; t < 4; ++t) {
            const __bf16* bh_p = ek_hi + (size_t)(t * 16 + mcol) * DIM + kb;
            const __bf16* bl_p = ek_lo + (size_t)(t * 16 + mcol) * DIM + kb;
            bf16x8 b0 = *(const bf16x8*)(bh_p);
            bf16x8 b1 = *(const bf16x8*)(bh_p + 16);
            bf16x8 c0 = *(const bf16x8*)(bl_p);
            bf16x8 c1 = *(const bf16x8*)(bl_p + 16);
            v16bf bh, bl;
            #pragma unroll
            for (int i = 0; i < 8; ++i) { bh[i] = b0[i]; bh[i+8] = b1[i]; bl[i] = c0[i]; bl[i+8] = c1[i]; }
            acc[t] = __builtin_amdgcn_wmma_f32_16x16x32_bf16(false, ah, false, bh, (short)0, acc[t], false, false);
            acc[t] = __builtin_amdgcn_wmma_f32_16x16x32_bf16(false, ah, false, bl, (short)0, acc[t], false, false);
            acc[t] = __builtin_amdgcn_wmma_f32_16x16x32_bf16(false, al, false, bh, (short)0, acc[t], false, false);
        }
    }
    // each lane holds half of row mcol's squares; combine halves (wave32 xor-16)
    nsq += __shfl_xor(nsq, 16, 32);
    const float zn = sqrtf(nsq);        // ||z[row_base + mcol]||

    // ================= similarity: divide, emit, stash =================
    float en[4];
    #pragma unroll
    for (int t = 0; t < 4; ++t) en[t] = enorm[t * 16 + mcol];
    #pragma unroll
    for (int r = 0; r < 8; ++r) {
        const int m = r + 8 * half;                 // C-tile row for this VGPR
        const float znr = __shfl(zn, m, 32);
        #pragma unroll
        for (int t = 0; t < 4; ++t) {
            float s = acc[t][r] / fmaxf(znr * en[t], EPSV);
            sim_out[(size_t)(row_base + m) * NEXP + t * 16 + mcol] = s;
            lds_w[wave][m][t * 16 + mcol] = s;
        }
    }
    __syncthreads();

    // ================= softmax over 64 experts (lane r owns row r) =================
    if (lane < 16) {
        float mx = -3.4e38f;
        #pragma unroll
        for (int e = 0; e < NEXP; ++e) mx = fmaxf(mx, lds_w[wave][lane][e]);
        float ssum = 0.f;
        #pragma unroll
        for (int e = 0; e < NEXP; ++e) ssum += __expf(lds_w[wave][lane][e] - mx);
        const float inv = 1.f / ssum;
        #pragma unroll
        for (int e = 0; e < NEXP; ++e)
            lds_w[wave][lane][e] = __expf(lds_w[wave][lane][e] - mx) * inv;
    }
    __syncthreads();

    // ================= GEMM2: weighted = W[16x64] @ E[64x2048] =================
    // A fragments (two K-steps of 32), split bf16
    v16bf wh[2], wl[2];
    #pragma unroll
    for (int s = 0; s < 2; ++s) {
        #pragma unroll
        for (int i = 0; i < 8; ++i) {
            float x0 = lds_w[wave][mcol][s * 32 + half * 8 + i];
            __bf16 h0 = (__bf16)x0;
            wh[s][i] = h0;  wl[s][i] = (__bf16)(x0 - (float)h0);
            float x1 = lds_w[wave][mcol][s * 32 + half * 8 + 16 + i];
            __bf16 h1 = (__bf16)x1;
            wh[s][i+8] = h1;  wl[s][i+8] = (__bf16)(x1 - (float)h1);
        }
    }

    for (int n = 0; n < DIM / 16; ++n) {
        const int col = n * 16 + mcol;
        const __bf16* bhb = ekt_hi + (size_t)col * NEXP;   // one full 128B row each
        const __bf16* blb = ekt_lo + (size_t)col * NEXP;
        v8f acc2 = (v8f){0.f,0.f,0.f,0.f,0.f,0.f,0.f,0.f};
        #pragma unroll
        for (int s = 0; s < 2; ++s) {
            const int base = s * 32 + half * 8;
            bf16x8 b0 = *(const bf16x8*)(bhb + base);
            bf16x8 b1 = *(const bf16x8*)(bhb + base + 16);
            bf16x8 c0 = *(const bf16x8*)(blb + base);
            bf16x8 c1 = *(const bf16x8*)(blb + base + 16);
            v16bf bh, bl;
            #pragma unroll
            for (int i = 0; i < 8; ++i) { bh[i] = b0[i]; bh[i+8] = b1[i]; bl[i] = c0[i]; bl[i+8] = c1[i]; }
            acc2 = __builtin_amdgcn_wmma_f32_16x16x32_bf16(false, wh[s], false, bh, (short)0, acc2, false, false);
            acc2 = __builtin_amdgcn_wmma_f32_16x16x32_bf16(false, wh[s], false, bl, (short)0, acc2, false, false);
            acc2 = __builtin_amdgcn_wmma_f32_16x16x32_bf16(false, wl[s], false, bh, (short)0, acc2, false, false);
        }
        #pragma unroll
        for (int r = 0; r < 8; ++r)
            wexp_out[(size_t)(row_base + r + 8 * half) * DIM + col] = acc2[r];
    }
}

extern "C" void kernel_launch(void* const* d_in, const int* in_sizes, int n_in,
                              void* d_out, int out_size, void* d_ws, size_t ws_size,
                              hipStream_t stream) {
    const float* z  = (const float*)d_in[0];
    const float* ek = (const float*)d_in[1];

    char* ws = (char*)d_ws;
    __bf16* ek_hi  = (__bf16*)(ws + WS_EK_HI);
    __bf16* ek_lo  = (__bf16*)(ws + WS_EK_LO);
    __bf16* ekt_hi = (__bf16*)(ws + WS_EKT_HI);
    __bf16* ekt_lo = (__bf16*)(ws + WS_EKT_LO);
    float*  enorm  = (float*)(ws + WS_ENORM);

    float* sim_out  = (float*)d_out;
    float* wexp_out = sim_out + (size_t)N_ROWS * NEXP;

    prep_split_kernel<<<(NEXP * DIM) / 256, 256, 0, stream>>>(ek, ek_hi, ek_lo, ekt_hi, ekt_lo);
    enorm_kernel<<<NEXP, 256, 0, stream>>>(ek, enorm);
    moe_cosine_kernel<<<N_ROWS / 128, 256, 0, stream>>>(z, ek_hi, ek_lo, ekt_hi, ekt_lo,
                                                        enorm, sim_out, wexp_out);
}
